// BatchTripletLoss_74337293959543
// MI455X (gfx1250) — matrix operand order repaired and checked by reading, not compiled, not run on hardware
//
#include <hip/hip_runtime.h>
#include <hip/hip_bf16.h>
#include <math.h>

// ---------------------------------------------------------------------------
// BatchTripletLoss on MI455X (gfx1250, wave32, WMMA + TDM)
//   batch [8192 x 1024] f32, positive [1024] f32 -> scalar f32
// bf16x3-split GEMM (batch @ batch^T) with fused row-max epilogue.
// v3: 6-arg tensor_load_to_lds (clang-23 arity), pass-major WMMA ordering,
//     TDM-driven double-buffered LDS staging (TENSORcnt async pipeline).
// ---------------------------------------------------------------------------

typedef __bf16 bf16_t;
typedef __attribute__((ext_vector_type(16))) bf16_t v16bf;
typedef __attribute__((ext_vector_type(4)))  bf16_t v4bf;
typedef __attribute__((ext_vector_type(8)))  float  v8f;
typedef __attribute__((ext_vector_type(4)))  unsigned int v4u;
typedef __attribute__((ext_vector_type(8)))  int    v8i;
typedef __attribute__((ext_vector_type(4)))  int    v4i;

#define NROWS   8192
#define DIM     1024
#define EPSQ    1e-6f
#define MARGINF 0.5f

#define BM 128          // i-rows per block (8 waves x 16)
#define BN 128          // j-cols per tile iteration
#define BK 32           // K step (bf16 WMMA K)
#define NK (DIM / BK)   // 32 k-steps per j-tile
#define JT_PER_BLK 8    // j-tiles per block
#define NSTEPS (JT_PER_BLK * NK)             // 256 pipeline steps
#define SPLITJ (NROWS / (BN * JT_PER_BLK))   // 8
#define LDS_STRIDE 40   // halves per LDS row (32 data + 8 pad)

#ifndef USE_TDM
#define USE_TDM 1
#endif
#if USE_TDM && defined(__has_builtin)
#if __has_builtin(__builtin_amdgcn_tensor_load_to_lds) && \
    __has_builtin(__builtin_amdgcn_s_wait_tensorcnt)
#define HAVE_TDM 1
#endif
#endif
#ifndef HAVE_TDM
#define HAVE_TDM 0
#endif

// ---- workspace layout (bytes) ---------------------------------------------
static constexpr size_t OFF_BHI  = 0;                                   // 16 MB
static constexpr size_t OFF_BLO  = (size_t)NROWS * DIM * sizeof(bf16_t);// +16 MB
static constexpr size_t OFF_SQ   = OFF_BLO + (size_t)NROWS * DIM * sizeof(bf16_t);
static constexpr size_t OFF_SUM  = OFF_SQ   + (size_t)NROWS * sizeof(float);
static constexpr size_t OFF_PD   = OFF_SUM  + (size_t)NROWS * sizeof(float);
static constexpr size_t OFF_VAL  = OFF_PD   + (size_t)NROWS * sizeof(float);
static constexpr size_t OFF_RMAX = OFF_VAL  + (size_t)NROWS * sizeof(float);

// ---------------------------------------------------------------------------
// Kernel 1: per-row stats + f32 -> bf16(hi) + bf16(lo residual) split
// ---------------------------------------------------------------------------
__global__ __launch_bounds__(256)
void prep_kernel(const float* __restrict__ batch, const float* __restrict__ pos,
                 bf16_t* __restrict__ bhi, bf16_t* __restrict__ blo,
                 float* __restrict__ sq, float* __restrict__ sums,
                 float* __restrict__ pdist, float* __restrict__ valid,
                 unsigned* __restrict__ rowmax)
{
    const int row = blockIdx.x;
    const int t   = threadIdx.x;
    const size_t base = (size_t)row * DIM + t * 4;

    const float4 x = *(const float4*)(batch + base);
    const float4 p = *(const float4*)(pos + t * 4);

    float xs[4] = {x.x, x.y, x.z, x.w};
    float ps[4] = {p.x, p.y, p.z, p.w};

    v4bf hv, lv;
    float s2 = 0.f, sm = 0.f, pd = 0.f;
    int eq = 1;
#pragma unroll
    for (int k = 0; k < 4; ++k) {
        const float v = xs[k];
        const bf16_t h = (bf16_t)v;            // RNE to bf16
        hv[k] = h;
        lv[k] = (bf16_t)(v - (float)h);        // residual
        s2 += v * v;
        sm += v;
        const float d = v - ps[k] + EPSQ;      // torch pairwise_distance eps
        pd += d * d;
        eq &= (v == ps[k]) ? 1 : 0;
    }
    *(v4bf*)(bhi + base) = hv;
    *(v4bf*)(blo + base) = lv;

#pragma unroll
    for (int off = 16; off > 0; off >>= 1) {
        s2 += __shfl_xor(s2, off, 32);
        sm += __shfl_xor(sm, off, 32);
        pd += __shfl_xor(pd, off, 32);
        eq &= __shfl_xor(eq, off, 32);
    }
    __shared__ float r2[8], rm[8], rp[8];
    __shared__ int   re[8];
    const int wid = t >> 5, lane = t & 31;
    if (lane == 0) { r2[wid] = s2; rm[wid] = sm; rp[wid] = pd; re[wid] = eq; }
    __syncthreads();
    if (t == 0) {
        float S2 = 0.f, SM = 0.f, PD = 0.f; int EQ = 1;
#pragma unroll
        for (int w = 0; w < 8; ++w) { S2 += r2[w]; SM += rm[w]; PD += rp[w]; EQ &= re[w]; }
        sq[row]     = S2;
        sums[row]   = SM;
        pdist[row]  = sqrtf(PD);
        valid[row]  = EQ ? 0.f : 1.f;
        rowmax[row] = 0u;                      // max(d2, 0) clamp built into init
    }
}

// ---------------------------------------------------------------------------
// TDM descriptor: load a 2-D tile (BK x BN bf16) from [NROWS x DIM] row-major
// global tensor into LDS, padding each 64 B row by 16 B -> LDS_STRIDE layout.
// ---------------------------------------------------------------------------
#if HAVE_TDM
__device__ __forceinline__
void tdm_load_tile(unsigned lds_off, const bf16_t* __restrict__ src,
                   int j0, int k0)
{
    const unsigned long long ga =
        (unsigned long long)(uintptr_t)(src + ((size_t)j0 * DIM + k0));
    v4u g0;
    g0.x = 1u;                                  // count=1, user descriptor
    g0.y = lds_off;                             // lds_addr (bytes)
    g0.z = (unsigned)(ga & 0xffffffffu);        // global_addr[31:0]
    g0.w = (unsigned)((ga >> 32) & 0x01ffffffu) // global_addr[56:32]
         | (2u << 30);                          // type=2 ("image")
    v8i g1;
    g1[0] = (1 << 16)      // data_size = 1 -> 2-byte elements
          | (1 << 20)      // pad_enable
          | (3 << 22)      // pad_interval: 16 DWORDs (64 B) between pads
          | (3 << 25);     // pad_amount: 4 DWORDs (16 B)
    g1[1] = (int)((unsigned)(DIM & 0xffff) << 16);   // tensor_dim0 lo16 at [79:48]
    g1[2] = (int)((unsigned)(NROWS & 0xffff) << 16); // dim0 hi=0 | tensor_dim1 lo16
    g1[3] = (int)(BK << 16);                         // dim1 hi=0 | tile_dim0=32
    g1[4] = BN;                                      // tile_dim1=128, tile_dim2=0
    g1[5] = DIM;                                     // tensor_dim0_stride lo32
    g1[6] = (int)((unsigned)(DIM & 0xffff) << 16);   // stride0 hi=0 | stride1 lo16
    g1[7] = 0;
    const v4i z4 = {0, 0, 0, 0};
    const v8i z8 = {0, 0, 0, 0, 0, 0, 0, 0};
    // clang-23 / therock arity: (g0, g1, g2, g3, g4, cpol)
    __builtin_amdgcn_tensor_load_to_lds(g0, g1, z4, z4, z8, 0);
}
#endif

// ---------------------------------------------------------------------------
// Kernel 2: bf16x3 GEMM (batch @ batch^T) with fused row-max epilogue.
// 256 threads (8 waves); wave w owns rows [i0+16w, i0+16w+16).
// Double-buffered LDS B tiles fed by TDM; pass-major WMMA over groups of 4
// sub-tiles so consecutive v_wmma never share an accumulator.
// ---------------------------------------------------------------------------
__global__ __launch_bounds__(256)
void gemm_max_kernel(const bf16_t* __restrict__ bhi, const bf16_t* __restrict__ blo,
                     const float* __restrict__ sq, const float* __restrict__ sums,
                     unsigned* __restrict__ rowmax)
{
    __shared__ __align__(16) bf16_t sB[2][2][BN][LDS_STRIDE]; // [buf][hi/lo][j][k]

    const int t    = threadIdx.x;
    const int lane = t & 31;
    const int wid  = t >> 5;
    const int half = lane >> 4;
    const int l16  = lane & 15;
    const int i0w  = blockIdx.y * BM + wid * 16;

    float sqi[8], smi[8];
#pragma unroll
    for (int m = 0; m < 8; ++m) {
        const int gi = i0w + m + 8 * half;
        sqi[m] = sq[gi];
        smi[m] = sums[gi];
    }

    const size_t arow = (size_t)(i0w + l16) * DIM;
    float rmax[8];
#pragma unroll
    for (int m = 0; m < 8; ++m) rmax[m] = 0.f;

    v8f acc[8];

    const int jbase = blockIdx.x * JT_PER_BLK;
#if HAVE_TDM
    // prologue: stage tile for step 0 into buffer 0
    if (wid == 0) {
        tdm_load_tile((unsigned)(uintptr_t)(void*)&sB[0][0][0][0], bhi, jbase * BN, 0);
        tdm_load_tile((unsigned)(uintptr_t)(void*)&sB[0][1][0][0], blo, jbase * BN, 0);
    }
#else
    // prologue: cooperative stage of step 0 into buffer 0
    {
        const int row = t >> 1;
        const int c   = (t & 1) * 16;
        const size_t g = (size_t)(jbase * BN + row) * DIM + c;
        const uint4* ph = (const uint4*)(bhi + g);
        *(uint4*)&sB[0][0][row][c]     = ph[0];
        *(uint4*)&sB[0][0][row][c + 8] = ph[1];
        const uint4* pl = (const uint4*)(blo + g);
        *(uint4*)&sB[0][1][row][c]     = pl[0];
        *(uint4*)&sB[0][1][row][c + 8] = pl[1];
    }
#endif

    int p = 0;
    for (int s = 0; s < NSTEPS; ++s) {
        const int jt = s >> 5;
        const int ks = s & (NK - 1);
        const int j0 = (jbase + jt) * BN;
        const int k0 = ks * BK;

#if HAVE_TDM
        if (wid == 0) __builtin_amdgcn_s_wait_tensorcnt(0); // buf[p] TDM done
        __syncthreads();                                    // publish; buf[p^1] free
        if (wid == 0 && (s + 1) < NSTEPS) {
            const int sn  = s + 1;
            const int jn0 = (jbase + (sn >> 5)) * BN;
            const int kn0 = (sn & (NK - 1)) * BK;
            tdm_load_tile((unsigned)(uintptr_t)(void*)&sB[p ^ 1][0][0][0], bhi, jn0, kn0);
            tdm_load_tile((unsigned)(uintptr_t)(void*)&sB[p ^ 1][1][0][0], blo, jn0, kn0);
        }
#else
        __syncthreads();                                    // buf[p] ready, buf[p^1] free
        uint4 nh0, nh1, nl0, nl1;
        const int row = t >> 1;
        const int c   = (t & 1) * 16;
        const bool have_next = (s + 1) < NSTEPS;
        if (have_next) {
            const int sn  = s + 1;
            const int jn0 = (jbase + (sn >> 5)) * BN;
            const int kn0 = (sn & (NK - 1)) * BK;
            const size_t g = (size_t)(jn0 + row) * DIM + kn0 + c;
            const uint4* ph = (const uint4*)(bhi + g);
            nh0 = ph[0]; nh1 = ph[1];
            const uint4* pl = (const uint4*)(blo + g);
            nl0 = pl[0]; nl1 = pl[1];
        }
#endif

        if (ks == 0) {
#pragma unroll
            for (int ns = 0; ns < 8; ++ns)
#pragma unroll
                for (int e = 0; e < 8; ++e) acc[ns][e] = 0.f;
        }

        // A fragments (ISA 16-bit A 16x32 layout)
        union AF { uint4 q[2]; v16bf v; } ah, al;
        {
            const size_t a0 = arow + k0 + half * 8;
            ah.q[0] = *(const uint4*)(bhi + a0);
            ah.q[1] = *(const uint4*)(bhi + a0 + 16);
            al.q[0] = *(const uint4*)(blo + a0);
            al.q[1] = *(const uint4*)(blo + a0 + 16);
        }

        // pass-major WMMA over two groups of 4 sub-tiles
#pragma unroll
        for (int g = 0; g < 2; ++g) {
            union BF { uint4 q[2]; v16bf v; } bh[4], bl[4];
#pragma unroll
            for (int q = 0; q < 4; ++q) {
                const int ns = g * 4 + q;
                const bf16_t* lh = &sB[p][0][ns * 16 + l16][half * 8];
                bh[q].q[0] = *(const uint4*)lh;
                bh[q].q[1] = *(const uint4*)(lh + 16);
                const bf16_t* ll = &sB[p][1][ns * 16 + l16][half * 8];
                bl[q].q[0] = *(const uint4*)ll;
                bl[q].q[1] = *(const uint4*)(ll + 16);
            }
#pragma unroll
            for (int q = 0; q < 4; ++q)
                acc[g * 4 + q] = __builtin_amdgcn_wmma_f32_16x16x32_bf16(
                    false, ah.v, false, bh[q].v, (short)0, acc[g * 4 + q], false, false);
#pragma unroll
            for (int q = 0; q < 4; ++q)
                acc[g * 4 + q] = __builtin_amdgcn_wmma_f32_16x16x32_bf16(
                    false, ah.v, false, bl[q].v, (short)0, acc[g * 4 + q], false, false);
#pragma unroll
            for (int q = 0; q < 4; ++q)
                acc[g * 4 + q] = __builtin_amdgcn_wmma_f32_16x16x32_bf16(
                    false, al.v, false, bh[q].v, (short)0, acc[g * 4 + q], false, false);
        }

#if !HAVE_TDM
        if (have_next) {
            *(uint4*)&sB[p ^ 1][0][row][c]     = nh0;
            *(uint4*)&sB[p ^ 1][0][row][c + 8] = nh1;
            *(uint4*)&sB[p ^ 1][1][row][c]     = nl0;
            *(uint4*)&sB[p ^ 1][1][row][c + 8] = nl1;
        }
#endif

        // fused epilogue at the end of each j-tile
        if (ks == NK - 1) {
#pragma unroll
            for (int ns = 0; ns < 8; ++ns) {
                const int gj    = j0 + ns * 16 + l16;
                const float sqj = sq[gj];
                const float smj = sums[gj];
#pragma unroll
                for (int m = 0; m < 8; ++m) {
                    const int gi = i0w + m + 8 * half;
                    const float d2 = sqi[m] + sqj - 2.0f * acc[ns][m]
                                   + 2.0f * EPSQ * (smi[m] - smj)
                                   + (float)DIM * EPSQ * EPSQ;
                    if (gi != gj) rmax[m] = fmaxf(rmax[m], d2);
                }
            }
        }
        p ^= 1;
    }

    // reduce max across the 16 lanes of each half (columns within tile)
#pragma unroll
    for (int m = 0; m < 8; ++m) {
        float v = rmax[m];
#pragma unroll
        for (int off = 1; off < 16; off <<= 1)
            v = fmaxf(v, __shfl_xor(v, off, 32));
        rmax[m] = v;
    }
    // merge across j-split blocks; values >= 0 so uint order == float order
    if (l16 == 0) {
#pragma unroll
        for (int m = 0; m < 8; ++m) {
            const int gi = i0w + m + 8 * half;
            atomicMax(rowmax + gi, __float_as_uint(rmax[m]));
        }
    }
}

// ---------------------------------------------------------------------------
// Kernel 3: final loss reduction
// ---------------------------------------------------------------------------
__global__ __launch_bounds__(256)
void finalize_kernel(const float* __restrict__ pdist, const float* __restrict__ valid,
                     const unsigned* __restrict__ rowmax, float* __restrict__ out)
{
    const int t = threadIdx.x;
    float sum = 0.f, cnt = 0.f;
    for (int i = t; i < NROWS; i += 256) {
        const float v  = valid[i];
        const float mn = sqrtf(__uint_as_float(rowmax[i]));
        const float loss = fmaxf(pdist[i] - mn + MARGINF, 0.f);
        sum += v * loss;
        cnt += v;
    }
#pragma unroll
    for (int off = 16; off > 0; off >>= 1) {
        sum += __shfl_xor(sum, off, 32);
        cnt += __shfl_xor(cnt, off, 32);
    }
    __shared__ float ss[8], sc[8];
    const int wid = t >> 5, lane = t & 31;
    if (lane == 0) { ss[wid] = sum; sc[wid] = cnt; }
    __syncthreads();
    if (t == 0) {
        float S = 0.f, C = 0.f;
#pragma unroll
        for (int w = 0; w < 8; ++w) { S += ss[w]; C += sc[w]; }
        out[0] = S / C;
    }
}

// ---------------------------------------------------------------------------
extern "C" void kernel_launch(void* const* d_in, const int* in_sizes, int n_in,
                              void* d_out, int out_size, void* d_ws, size_t ws_size,
                              hipStream_t stream)
{
    const float* batch = (const float*)d_in[0];   // [8192, 1024]
    const float* pos   = (const float*)d_in[1];   // [1024]
    float* out = (float*)d_out;

    char* ws = (char*)d_ws;
    bf16_t*   bhi    = (bf16_t*)(ws + OFF_BHI);
    bf16_t*   blo    = (bf16_t*)(ws + OFF_BLO);
    float*    sq     = (float*)(ws + OFF_SQ);
    float*    sums   = (float*)(ws + OFF_SUM);
    float*    pdist  = (float*)(ws + OFF_PD);
    float*    valid  = (float*)(ws + OFF_VAL);
    unsigned* rowmax = (unsigned*)(ws + OFF_RMAX);

    prep_kernel<<<NROWS, 256, 0, stream>>>(batch, pos, bhi, blo, sq, sums,
                                           pdist, valid, rowmax);

    dim3 grid(SPLITJ, NROWS / BM);   // (8, 64) = 512 blocks
    gemm_max_kernel<<<grid, 256, 0, stream>>>(bhi, blo, sq, sums, rowmax);

    finalize_kernel<<<1, 256, 0, stream>>>(pdist, valid, rowmax, out);
}
